// SymbolicPlanner_4131758539379
// MI455X (gfx1250) — matrix operand order repaired and checked by reading, not compiled, not run on hardware
//
#include <hip/hip_runtime.h>
#include <hip/hip_bf16.h>

// ---------------------------------------------------------------------------
// Problem constants
// ---------------------------------------------------------------------------
#define BB    65536   // batch
#define XDIM  362
#define ZDIM  100
#define NEXP  3

// Padded dims: every layer-output (N) padded to a multiple of 64 so each
// workgroup panel is exactly 4 WMMA N-tiles; K dims (= previous N, or 384 for
// the input) are then automatically multiples of 32 for 16x16x32 bf16 WMMA.
#define XP    384     // input K: 362 -> 384
#define H1P   256     // 256
#define H2P   256     // 200 -> 256
#define ZP    128     // 100 -> 128
#define E1P   192     // 150 -> 192
#define E2P   256     // 250 -> 256
#define E3P   320     // 300 -> 320
#define E4P   384     // 362 -> 384

typedef __attribute__((ext_vector_type(16))) __bf16 v16bf;
typedef __attribute__((ext_vector_type(8)))  float  v8f;
typedef __attribute__((ext_vector_type(4)))  unsigned int u32x4;

union FragBF {
    v16bf v;
    u32x4 u[2];
};

// round-to-nearest-even fp32 -> bf16 (bit-level)
__device__ __forceinline__ __bf16 f2bf(float f) {
    union { float f; unsigned u; } c; c.f = f;
    unsigned r = c.u + 0x7FFFu + ((c.u >> 16) & 1u);
    unsigned short h = (unsigned short)(r >> 16);
    return __builtin_bit_cast(__bf16, h);
}

// CDNA5 async global->LDS copy (16B per lane, ASYNCcnt-tracked)
__device__ __forceinline__ void async_copy_b128(unsigned lds_off, const __bf16* gsrc) {
    asm volatile("global_load_async_to_lds_b128 %0, %1, off"
                 :: "v"(lds_off), "v"(gsrc) : "memory");
}
__device__ __forceinline__ void wait_async0() {
    asm volatile("s_wait_asynccnt 0" ::: "memory");
}

// ---------------------------------------------------------------------------
// Pad/convert batch_x (fp32 [B,362]) -> bf16 [B,384], zero padded cols
// ---------------------------------------------------------------------------
__global__ __launch_bounds__(256)
void k_conv_x(const float* __restrict__ x, __bf16* __restrict__ xb) {
    size_t i = (size_t)blockIdx.x * blockDim.x + threadIdx.x;   // over B*XP
    size_t row = i / XP;
    int    col = (int)(i % XP);
    if (row >= BB) return;
    float v = (col < XDIM) ? x[row * XDIM + col] : 0.0f;
    xb[i] = f2bf(v);
}

// ---------------------------------------------------------------------------
// Convert W [K,N] fp32 row-major -> W^T bf16 [Npad,Kpad], zero padded
// ---------------------------------------------------------------------------
__global__ __launch_bounds__(256)
void k_conv_w(const float* __restrict__ W, __bf16* __restrict__ Wt,
              int K, int N, int Kpad, int Npad) {
    size_t i = (size_t)blockIdx.x * blockDim.x + threadIdx.x;   // over Npad*Kpad
    if (i >= (size_t)Npad * Kpad) return;
    int n = (int)(i / Kpad);
    int k = (int)(i % Kpad);
    float v = (n < N && k < K) ? W[(size_t)k * N + n] : 0.0f;
    Wt[i] = f2bf(v);
}

// ---------------------------------------------------------------------------
// GEMM: C[B,Npad] = act(A[B,KPAD] @ W + bias), bf16 in/out, f32 accum.
// Block = 8 wave32s; block tile = 128 rows x 64 cols.
// Each wave: one 16-row strip, 4 accumulator tiles (A fragment reused 4x).
// Weight panel (64 x 32 K-slice, 4 KB) staged in LDS via async global->LDS,
// double buffered; K loop fully unrolled (KPAD is a template constant).
// ACT: 0 = relu, 1 = sigmoid
// ---------------------------------------------------------------------------
template <int KPAD, int ACT>
__global__ __launch_bounds__(256)
void k_gemm(const __bf16* __restrict__ A, const __bf16* __restrict__ Wt,
            const float* __restrict__ bias, __bf16* __restrict__ C,
            int Npad, int Nreal) {
    __shared__ __align__(16) __bf16 Bp[2][64 * 32];

    const int tid  = threadIdx.x;
    const int lane = tid & 31;
    const int wave = tid >> 5;
    const int r    = lane & 15;
    const int hi   = lane >> 4;

    const int n0 = blockIdx.x * 64;
    const int m0 = blockIdx.y * 128 + wave * 16;

    // async-copy mapping: thread -> (panel row, 8-element K chunk)
    const int prow = tid >> 2;          // 0..63  (W^T row = output col)
    const int pk   = (tid & 3) * 8;     // 0,8,16,24
    const __bf16* Wrow = Wt + (size_t)(n0 + prow) * KPAD + pk;
    const unsigned ldsW0 = (unsigned)(size_t)(void*)&Bp[0][prow * 32 + pk];
    const unsigned ldsW1 = (unsigned)(size_t)(void*)&Bp[1][prow * 32 + pk];

    // A fragment base: lane = row m0+r; halves 0..7 -> K+hi*16, halves 8..15 -> +8
    const __bf16* Arow = A + (size_t)(m0 + r) * KPAD + (hi ? 16 : 0);
    const __bf16* Bl0  = &Bp[0][r * 32 + hi * 16];
    const __bf16* Bl1  = &Bp[1][r * 32 + hi * 16];

    // preload panel 0
    async_copy_b128(ldsW0, Wrow);
    wait_async0();
    __syncthreads();

    v8f acc[4] = {};
#pragma unroll
    for (int k0 = 0; k0 < KPAD; k0 += 32) {
        constexpr int NSTEP = KPAD / 32;
        const int  step = k0 >> 5;
        const int  cur  = step & 1;
        const bool more = (step + 1) < NSTEP;
        if (more)
            async_copy_b128(cur ? ldsW0 : ldsW1, Wrow + k0 + 32);

        FragBF a;
        a.u[0] = *(const u32x4*)(Arow + k0);
        a.u[1] = *(const u32x4*)(Arow + k0 + 8);

        // load all 4 B fragments into independent regs -> one dscnt wait,
        // then 4 independent back-to-back WMMAs (XDL co-execution)
        const __bf16* bl = cur ? Bl1 : Bl0;
        FragBF b0, b1, b2, b3;
        b0.u[0] = *(const u32x4*)(bl);
        b0.u[1] = *(const u32x4*)(bl + 8);
        b1.u[0] = *(const u32x4*)(bl + 16 * 32);
        b1.u[1] = *(const u32x4*)(bl + 16 * 32 + 8);
        b2.u[0] = *(const u32x4*)(bl + 32 * 32);
        b2.u[1] = *(const u32x4*)(bl + 32 * 32 + 8);
        b3.u[0] = *(const u32x4*)(bl + 48 * 32);
        b3.u[1] = *(const u32x4*)(bl + 48 * 32 + 8);

        acc[0] = __builtin_amdgcn_wmma_f32_16x16x32_bf16(
            false, a.v, false, b0.v, (short)0, acc[0], false, false);
        acc[1] = __builtin_amdgcn_wmma_f32_16x16x32_bf16(
            false, a.v, false, b1.v, (short)0, acc[1], false, false);
        acc[2] = __builtin_amdgcn_wmma_f32_16x16x32_bf16(
            false, a.v, false, b2.v, (short)0, acc[2], false, false);
        acc[3] = __builtin_amdgcn_wmma_f32_16x16x32_bf16(
            false, a.v, false, b3.v, (short)0, acc[3], false, false);

        if (more) wait_async0();
        __syncthreads();
    }

    // Epilogue: C layout — vgpr i: row m0+i+hi*8, col n0+t*16+(lane&15)
#pragma unroll
    for (int t = 0; t < 4; ++t) {
        const int col = n0 + t * 16 + r;
        const float bv = (col < Nreal) ? bias[col] : 0.0f;
#pragma unroll
        for (int i = 0; i < 8; ++i) {
            const int row = m0 + i + (hi << 3);
            float v = 0.0f;
            if (col < Nreal) {
                v = acc[t][i] + bv;
                if (ACT == 0) v = v > 0.0f ? v : 0.0f;
                else          v = 1.0f / (1.0f + __expf(-v));
            }
            C[(size_t)row * Npad + col] = f2bf(v);
        }
    }
}

// ---------------------------------------------------------------------------
// Final expert layer: same GEMM structure (K = E3P), epilogue fuses routing
// gather + squared-error accumulation (col 0 vs cols 1..) + LDS reduction.
// ---------------------------------------------------------------------------
__global__ __launch_bounds__(256)
void k_expert_out_loss(const __bf16* __restrict__ A, const __bf16* __restrict__ Wt,
                       const float* __restrict__ bias,
                       const float* __restrict__ bx, const float* __restrict__ bnext,
                       const int* __restrict__ ba, int e,
                       float* __restrict__ accum) {
    __shared__ __align__(16) __bf16 Bp[2][64 * 32];
    __shared__ float red0[256], red1[256];

    const int tid  = threadIdx.x;
    const int lane = tid & 31;
    const int wave = tid >> 5;
    const int r    = lane & 15;
    const int hi   = lane >> 4;

    const int n0 = blockIdx.x * 64;
    const int m0 = blockIdx.y * 128 + wave * 16;

    const int prow = tid >> 2;
    const int pk   = (tid & 3) * 8;
    const __bf16* Wrow = Wt + (size_t)(n0 + prow) * E3P + pk;
    const unsigned ldsW0 = (unsigned)(size_t)(void*)&Bp[0][prow * 32 + pk];
    const unsigned ldsW1 = (unsigned)(size_t)(void*)&Bp[1][prow * 32 + pk];
    const __bf16* Arow = A + (size_t)(m0 + r) * E3P + (hi ? 16 : 0);
    const __bf16* Bl0  = &Bp[0][r * 32 + hi * 16];
    const __bf16* Bl1  = &Bp[1][r * 32 + hi * 16];

    async_copy_b128(ldsW0, Wrow);
    wait_async0();
    __syncthreads();

    v8f acc[4] = {};
#pragma unroll
    for (int k0 = 0; k0 < E3P; k0 += 32) {
        constexpr int NSTEP = E3P / 32;
        const int  step = k0 >> 5;
        const int  cur  = step & 1;
        const bool more = (step + 1) < NSTEP;
        if (more)
            async_copy_b128(cur ? ldsW0 : ldsW1, Wrow + k0 + 32);

        FragBF a;
        a.u[0] = *(const u32x4*)(Arow + k0);
        a.u[1] = *(const u32x4*)(Arow + k0 + 8);

        const __bf16* bl = cur ? Bl1 : Bl0;
        FragBF b0, b1, b2, b3;
        b0.u[0] = *(const u32x4*)(bl);
        b0.u[1] = *(const u32x4*)(bl + 8);
        b1.u[0] = *(const u32x4*)(bl + 16 * 32);
        b1.u[1] = *(const u32x4*)(bl + 16 * 32 + 8);
        b2.u[0] = *(const u32x4*)(bl + 32 * 32);
        b2.u[1] = *(const u32x4*)(bl + 32 * 32 + 8);
        b3.u[0] = *(const u32x4*)(bl + 48 * 32);
        b3.u[1] = *(const u32x4*)(bl + 48 * 32 + 8);

        acc[0] = __builtin_amdgcn_wmma_f32_16x16x32_bf16(
            false, a.v, false, b0.v, (short)0, acc[0], false, false);
        acc[1] = __builtin_amdgcn_wmma_f32_16x16x32_bf16(
            false, a.v, false, b1.v, (short)0, acc[1], false, false);
        acc[2] = __builtin_amdgcn_wmma_f32_16x16x32_bf16(
            false, a.v, false, b2.v, (short)0, acc[2], false, false);
        acc[3] = __builtin_amdgcn_wmma_f32_16x16x32_bf16(
            false, a.v, false, b3.v, (short)0, acc[3], false, false);

        if (more) wait_async0();
        __syncthreads();
    }

    float s0 = 0.0f, s1 = 0.0f;
#pragma unroll
    for (int t = 0; t < 4; ++t) {
        const int col = n0 + t * 16 + r;
        const float bv = (col < XDIM) ? bias[col] : 0.0f;
#pragma unroll
        for (int i = 0; i < 8; ++i) {
            const int row = m0 + i + (hi << 3);
            if (col < XDIM && ba[row] == e) {
                const size_t idx = (size_t)row * XDIM + col;
                const float p = acc[t][i] + bv;
                const float d = bnext[idx] - bx[idx];
                const float q = (p - d) * (p - d);
                if (col == 0) s0 += q; else s1 += q;
            }
        }
    }

    red0[tid] = s0;
    red1[tid] = s1;
    __syncthreads();
    for (int stride = 128; stride > 0; stride >>= 1) {
        if (tid < stride) {
            red0[tid] += red0[tid + stride];
            red1[tid] += red1[tid + stride];
        }
        __syncthreads();
    }
    if (tid == 0) {
        atomicAdd(&accum[0], red0[0]);
        atomicAdd(&accum[1], red1[0]);
    }
}

__global__ void k_zero_acc(float* a) { a[0] = 0.0f; a[1] = 0.0f; }

__global__ void k_finalize(const float* __restrict__ acc, float* __restrict__ out) {
    out[0] = acc[1] / ((float)BB * (float)(XDIM - 1)) + acc[0] / (float)BB;
}

// ---------------------------------------------------------------------------
extern "C" void kernel_launch(void* const* d_in, const int* in_sizes, int n_in,
                              void* d_out, int out_size, void* d_ws, size_t ws_size,
                              hipStream_t stream) {
    const float* bx    = (const float*)d_in[0];
    const float* bnext = (const float*)d_in[1];
    const int*   ba    = (const int*)  d_in[2];
    const float* W1 = (const float*)d_in[3];  const float* b1 = (const float*)d_in[4];
    const float* W2 = (const float*)d_in[5];  const float* b2 = (const float*)d_in[6];
    const float* W3 = (const float*)d_in[7];  const float* b3 = (const float*)d_in[8];
    const float* eW1 = (const float*)d_in[9];  const float* eb1 = (const float*)d_in[10];
    const float* eW2 = (const float*)d_in[11]; const float* eb2 = (const float*)d_in[12];
    const float* eW3 = (const float*)d_in[13]; const float* eb3 = (const float*)d_in[14];
    const float* eW4 = (const float*)d_in[15]; const float* eb4 = (const float*)d_in[16];

    char* ws = (char*)d_ws;
    size_t off = 0;
    auto alloc = [&](size_t bytes) -> char* {
        char* p = ws + off;
        off += (bytes + 255) & ~(size_t)255;
        return p;
    };

    float*  acc = (float*)alloc(2 * sizeof(float));
    __bf16* Xbf = (__bf16*)alloc((size_t)BB * XP  * 2);
    __bf16* H1  = (__bf16*)alloc((size_t)BB * H1P * 2);
    __bf16* H2  = (__bf16*)alloc((size_t)BB * H2P * 2);
    __bf16* Zb  = (__bf16*)alloc((size_t)BB * ZP  * 2);
    // expert intermediates alias freed encoder buffers (sequential launches)
    __bf16* E1  = H2;    // B*192 <= B*256
    __bf16* E2  = H1;    // B*256 == B*256
    __bf16* E3  = Xbf;   // B*320 <= B*384
    __bf16* W1t = (__bf16*)alloc((size_t)H1P * XP  * 2);
    __bf16* W2t = (__bf16*)alloc((size_t)H2P * H1P * 2);
    __bf16* W3t = (__bf16*)alloc((size_t)ZP  * H2P * 2);
    __bf16* e1t = (__bf16*)alloc((size_t)E1P * ZP  * 2);
    __bf16* e2t = (__bf16*)alloc((size_t)E2P * E1P * 2);
    __bf16* e3t = (__bf16*)alloc((size_t)E3P * E2P * 2);
    __bf16* e4t = (__bf16*)alloc((size_t)E4P * E3P * 2);

    const int T = 256;
    auto blk = [](size_t n, int t) { return (unsigned)((n + t - 1) / t); };

    k_zero_acc<<<1, 1, 0, stream>>>(acc);

    // Convert inputs / encoder weights
    k_conv_x<<<blk((size_t)BB * XP, T), T, 0, stream>>>(bx, Xbf);
    k_conv_w<<<blk((size_t)H1P * XP,  T), T, 0, stream>>>(W1, W1t, XDIM, 256, XP,  H1P);
    k_conv_w<<<blk((size_t)H2P * H1P, T), T, 0, stream>>>(W2, W2t, 256,  200, H1P, H2P);
    k_conv_w<<<blk((size_t)ZP  * H2P, T), T, 0, stream>>>(W3, W3t, 200, ZDIM, H2P, ZP);

    const unsigned rowPanels = BB / 128;   // 512
    // Encoder
    k_gemm<XP,  0><<<dim3(H1P / 64, rowPanels), T, 0, stream>>>(Xbf, W1t, b1, H1, H1P, 256);
    k_gemm<H1P, 0><<<dim3(H2P / 64, rowPanels), T, 0, stream>>>(H1,  W2t, b2, H2, H2P, 200);
    k_gemm<H2P, 1><<<dim3(ZP  / 64, rowPanels), T, 0, stream>>>(H2,  W3t, b3, Zb, ZP,  ZDIM);

    // Experts (all E, gather fused into final layer's loss epilogue)
    for (int e = 0; e < NEXP; ++e) {
        k_conv_w<<<blk((size_t)E1P * ZP,  T), T, 0, stream>>>(eW1 + (size_t)e * ZDIM * 150, e1t, ZDIM, 150, ZP,  E1P);
        k_conv_w<<<blk((size_t)E2P * E1P, T), T, 0, stream>>>(eW2 + (size_t)e * 150 * 250,  e2t, 150, 250, E1P, E2P);
        k_conv_w<<<blk((size_t)E3P * E2P, T), T, 0, stream>>>(eW3 + (size_t)e * 250 * 300,  e3t, 250, 300, E2P, E3P);
        k_conv_w<<<blk((size_t)E4P * E3P, T), T, 0, stream>>>(eW4 + (size_t)e * 300 * XDIM, e4t, 300, XDIM, E3P, E4P);

        k_gemm<ZP,  0><<<dim3(E1P / 64, rowPanels), T, 0, stream>>>(Zb, e1t, eb1 + (size_t)e * 150, E1, E1P, 150);
        k_gemm<E1P, 0><<<dim3(E2P / 64, rowPanels), T, 0, stream>>>(E1, e2t, eb2 + (size_t)e * 250, E2, E2P, 250);
        k_gemm<E2P, 0><<<dim3(E3P / 64, rowPanels), T, 0, stream>>>(E2, e3t, eb3 + (size_t)e * 300, E3, E3P, 300);
        k_expert_out_loss<<<dim3(E4P / 64, rowPanels), T, 0, stream>>>(
            E3, e4t, eb4 + (size_t)e * XDIM, bx, bnext, ba, e, acc);
    }

    k_finalize<<<1, 1, 0, stream>>>(acc, (float*)d_out);
}
